// Hyena2DConv_52158082843344
// MI455X (gfx1250) — matrix operand order repaired and checked by reading, not compiled, not run on hardware
//
#include <hip/hip_runtime.h>
#include <hip/hip_bf16.h>
#include <stdint.h>

typedef _Float16 v16h __attribute__((ext_vector_type(16)));
typedef _Float16 h8   __attribute__((ext_vector_type(8)));
typedef float    v8f  __attribute__((ext_vector_type(8)));
typedef int      gv4i __attribute__((vector_size(4 * sizeof(int))));  // int4 per builtin sig

#define DIMC   192
#define KS     55
#define HW     56
#define KPADD  64   // padded filter rows / taps (55 -> 64, zero filled)
#define NRSTG  4    // rows staged per barrier period

// ---------------------------------------------------------------------------
// Kernel 0: zero the padded f16 filter workspace [192][64][64]
// ---------------------------------------------------------------------------
__global__ void hyena_zero_ws(uint32_t* __restrict__ w, int n) {
  int i = blockIdx.x * blockDim.x + threadIdx.x;
  if (i < n) w[i] = 0u;
}

// ---------------------------------------------------------------------------
// Kernel 1: filter generation (tiny MLP per spatial position, 55x55=3025 px)
// ---------------------------------------------------------------------------
__global__ __launch_bounds__(256)
void hyena_gen_filter(const float* __restrict__ z,
                      const float* __restrict__ W1, const float* __restrict__ b1,
                      const float* __restrict__ f1,
                      const float* __restrict__ W2, const float* __restrict__ b2,
                      const float* __restrict__ f2,
                      const float* __restrict__ W3,
                      const float* __restrict__ deltas,
                      const float* __restrict__ shiftv,
                      const float* __restrict__ dist,
                      _Float16* __restrict__ kf)
{
  int p = blockIdx.x * blockDim.x + threadIdx.x;
  if (p >= KS * KS) return;
  const int a = p / KS, t = p % KS;

  float zv[32];
  #pragma unroll
  for (int cc = 0; cc < 32; ++cc) zv[cc] = z[cc * (KS * KS) + p];

  float h1[64];
  for (int o = 0; o < 64; ++o) {
    float s = b1[o];
    #pragma unroll
    for (int cc = 0; cc < 32; ++cc) s += W1[o * 32 + cc] * zv[cc];
    h1[o] = sinf(f1[o] * s);
  }
  float h2[64];
  for (int o = 0; o < 64; ++o) {
    float s = b2[o];
    #pragma unroll
    for (int cc = 0; cc < 64; ++cc) s += W2[o * 64 + cc] * h1[cc];
    h2[o] = sinf(f2[o] * s);
  }
  const float d = dist[p];
  for (int c = 0; c < DIMC; ++c) {
    float s = 0.f;
    #pragma unroll
    for (int o = 0; o < 64; ++o) s += W3[c * 64 + o] * h2[o];
    const float val = s * (expf(-d * fabsf(deltas[c])) + shiftv[c]);
    kf[(c * KPADD + a) * KPADD + t] = (_Float16)val;
  }
}

// ---------------------------------------------------------------------------
// Kernel 2: depthwise 55x55 conv via WMMA f16->f32 (Toeplitz formulation).
// A = 16 filter rows (aligned b128 LDS loads), B = Toeplitz reversed x row
// (parity-selected aligned b32 LDS loads). D: i = i0+rv+8*(lane/16),
// j = j0 + lane%16. 4 rows staged per barrier, double-buffered.
// ---------------------------------------------------------------------------
__global__ __launch_bounds__(128)
void hyena_conv_wmma(const float* __restrict__ x,
                     const _Float16* __restrict__ kf,
                     const float* __restrict__ biasv,
                     float* __restrict__ out)
{
  // kpad rows: [0,16) zero guard (a<0), [16,80) = filter rows 0..63,
  // [80,96) zero guard.
  __shared__ __align__(16) _Float16 kpad[96 * KPADD];
  __shared__ __align__(16) _Float16 xrevA[2][NRSTG][128];  // [u] = x[r, 90-u]
  __shared__ __align__(16) _Float16 xrevO[2][NRSTG][128];  // [u] = x[r, 89-u]

  const int tid  = threadIdx.x;
  const int lane = tid & 31;
  const int wid  = tid >> 5;
  const int Ml   = lane & 15;   // A: filter-row offset; B/D: out-col offset
  const int hl   = lane >> 4;
  const int j0   = wid << 4;    // column tile base per wave

  const int it = blockIdx.x & 3;
  const int bc = blockIdx.x >> 2;
  const int c  = bc % DIMC;
  const int b  = bc / DIMC;
  const int i0 = it << 4;

  // ---- zero ONLY the guard rows (no overlap with async-copied region) ----
  {
    uint32_t* g0 = (uint32_t*)kpad;                    // rows [0,16)
    uint32_t* g1 = (uint32_t*)(kpad + 80 * KPADD);     // rows [80,96)
    #pragma unroll
    for (int q = 0; q < 4; ++q) {                      // 512 words each
      g0[tid + 128 * q] = 0u;
      g1[tid + 128 * q] = 0u;
    }
  }

  // ---- stage filter rows 0..63 into kpad rows 16..79 (async -> LDS) ----
#if __has_builtin(__builtin_amdgcn_global_load_async_to_lds_b128)
  {
    const char* gsrc = (const char*)(kf + (size_t)c * KPADD * KPADD);
    char*       ldst = (char*)(kpad + 16 * KPADD);
    #pragma unroll
    for (int q = 0; q < 4; ++q) {                      // 512 x 16B = 8 KB
      const int idx = tid + 128 * q;
      __builtin_amdgcn_global_load_async_to_lds_b128(
          (gv4i*)(gsrc + (size_t)idx * 16),
          (gv4i*)(ldst + (size_t)idx * 16), 0, 0);
    }
  }
  #if __has_builtin(__builtin_amdgcn_s_wait_asynccnt)
    __builtin_amdgcn_s_wait_asynccnt(0);
  #else
    asm volatile("s_wait_asynccnt 0x0" ::: "memory");
  #endif
#else
  {
    const uint32_t* kg = (const uint32_t*)(kf + (size_t)c * KPADD * KPADD);
    uint32_t*       kl = (uint32_t*)(kpad + 16 * KPADD);
    for (int idx = tid; idx < KPADD * KPADD / 2; idx += 128) kl[idx] = kg[idx];
  }
#endif

  const float* xb = x + (size_t)(b * DIMC + c) * HW * HW;

  int r0 = i0 - 27; if (r0 < 0) r0 = 0;
  int r1 = i0 + 42; if (r1 > HW - 1) r1 = HW - 1;

  // cooperative group staging: thread `tid` covers element u = tid of the
  // reversed row; the shifted copy reuses the same loaded value.
  auto stage_group = [&](int buf, int rstart) {
    #pragma unroll
    for (int rr = 0; rr < NRSTG; ++rr) {
      const int r = rstart + rr;                 // block-uniform condition
      if (r <= r1) {
        const float* xsrc = xb + r * HW;
        const int u  = tid;
        const int cA = 90 - u;
        const float v = (cA >= 0 && cA < HW) ? xsrc[cA] : 0.f;
        const _Float16 hv = (_Float16)v;
        xrevA[buf][rr][u] = hv;
        if (u >= 1) xrevO[buf][rr][u - 1] = hv;  // x[r,89-(u-1)] = x[r,90-u]
        if (u == 0) xrevO[buf][rr][127] = (_Float16)0.f;
      }
    }
  };

  // prologue: stage first group into buffer 0
  stage_group(0, r0);
  __syncthreads();

  v8f acc = {0.f, 0.f, 0.f, 0.f, 0.f, 0.f, 0.f, 0.f};

  const int u0 = 63 - j0 - Ml;                   // B[t,n]: xrev[u0 + t]

  int sb = 0;
  for (int base = r0; base <= r1; base += NRSTG, sb ^= 1) {
    // stage next group while computing the current one
    if (base + NRSTG <= r1) {
      __builtin_prefetch(xb + (base + 2 * NRSTG) * HW, 0, 1);
      stage_group(sb ^ 1, base + NRSTG);
    }

    #pragma unroll
    for (int rr = 0; rr < NRSTG; ++rr) {
      const int r = base + rr;                   // block-uniform condition
      if (r <= r1) {
        const uint32_t* xwA = (const uint32_t*)xrevA[sb][rr];
        const uint32_t* xwO = (const uint32_t*)xrevO[sb][rr];
        const int arow0 = i0 + 27 - r;           // filter row for m=0
        const _Float16* krow = kpad + (16 + arow0 + Ml) * KPADD;

        #pragma unroll
        for (int ks = 0; ks < 2; ++ks) {         // taps [0,32) and [32,64)
          // A (16x32 f16, filter): two aligned 16B runs per lane
          const _Float16* ka = krow + 32 * ks + 16 * hl;
          h8 lo = *(const h8*)(ka);
          h8 hi = *(const h8*)(ka + 8);
          v16h am = __builtin_shufflevector(lo, hi,
              0, 1, 2, 3, 4, 5, 6, 7, 8, 9, 10, 11, 12, 13, 14, 15);

          // B (32x16 f16, Toeplitz x): contiguous 16 halves per lane,
          // parity-selected copy so all loads are aligned b32
          const int base_u = u0 + 32 * ks + 16 * hl;
          const uint32_t* bsrc = (base_u & 1) ? xwO : xwA;
          const int w0 = base_u >> 1;
          union { uint32_t w[8]; v16h h; } bu;
          #pragma unroll
          for (int e = 0; e < 8; ++e) bu.w[e] = bsrc[w0 + e];

          acc = __builtin_amdgcn_wmma_f32_16x16x32_f16(
              /*neg_a=*/false, am, /*neg_b=*/false, bu.h,
              /*c_mod=*/(short)0, acc, /*reuse_a=*/false, /*reuse_b=*/false);
        }
      }
    }
    __syncthreads();
  }

  // ---- epilogue: D[rv][lane]: i = i0 + rv + 8*hl, j = j0 + lane%16 ----
  const int j = j0 + Ml;
  if (j < HW) {
    const float bcv = biasv[c];
    const size_t obase = (size_t)(b * DIMC + c) * HW * HW;
    #pragma unroll
    for (int rv = 0; rv < 8; ++rv) {
      const int i = i0 + rv + 8 * hl;
      if (i < HW) {
        const size_t off = obase + (size_t)i * HW + j;
        out[off] = acc[rv] + bcv * x[off];
      }
    }
  }
}

// ---------------------------------------------------------------------------
// Launcher
// in order: 0:x 1:z 2:W1 3:b1 4:f1 5:W2 6:b2 7:f2 8:W3 9:deltas 10:shift
//           11:bias 12:distance_map
// ---------------------------------------------------------------------------
extern "C" void kernel_launch(void* const* d_in, const int* in_sizes, int n_in,
                              void* d_out, int out_size, void* d_ws, size_t ws_size,
                              hipStream_t stream) {
  const float* x      = (const float*)d_in[0];
  const float* z      = (const float*)d_in[1];
  const float* W1     = (const float*)d_in[2];
  const float* b1     = (const float*)d_in[3];
  const float* f1     = (const float*)d_in[4];
  const float* W2     = (const float*)d_in[5];
  const float* b2     = (const float*)d_in[6];
  const float* f2     = (const float*)d_in[7];
  const float* W3     = (const float*)d_in[8];
  const float* deltas = (const float*)d_in[9];
  const float* shiftv = (const float*)d_in[10];
  const float* biasv  = (const float*)d_in[11];
  const float* dist   = (const float*)d_in[12];
  float* out          = (float*)d_out;

  _Float16* kf = (_Float16*)d_ws;                 // [192][64][64] f16 = 1.5 MB

  const int nwords = DIMC * KPADD * KPADD / 2;    // 393216 u32
  hyena_zero_ws<<<(nwords + 255) / 256, 256, 0, stream>>>((uint32_t*)kf, nwords);

  hyena_gen_filter<<<(KS * KS + 255) / 256, 256, 0, stream>>>(
      z, W1, b1, f1, W2, b2, f2, W3, deltas, shiftv, dist, kf);

  hyena_conv_wmma<<<64 * DIMC * 4, 128, 0, stream>>>(x, kf, biasv, out);
}
